// Model_37177236914661
// MI455X (gfx1250) — matrix operand order repaired and checked by reading, not compiled, not run on hardware
//
#include <hip/hip_runtime.h>

// ---------------------------------------------------------------------------
// Types for WMMA bf16 (codegen-confirmed builtin signature)
// ---------------------------------------------------------------------------
typedef __attribute__((ext_vector_type(16))) __bf16 v16bf;
typedef __attribute__((ext_vector_type(8)))  float  v8f;

struct U32x8  { uint4 lo; uint4 hi; };
static_assert(sizeof(U32x8) == 32, "");

__device__ __forceinline__ unsigned short f2bf(float f) {
    unsigned u = __builtin_bit_cast(unsigned, f);
    u += 0x7FFFu + ((u >> 16) & 1u);          // round-to-nearest-even
    return (unsigned short)(u >> 16);
}

// ---------------------------------------------------------------------------
// Utility kernels
// ---------------------------------------------------------------------------
__global__ void zero_f32(float* __restrict__ p, long long n) {
    long long stride = (long long)gridDim.x * blockDim.x;
    for (long long i = (long long)blockIdx.x * blockDim.x + threadIdx.x; i < n; i += stride)
        p[i] = 0.0f;
}

__global__ void count_edges(const int* __restrict__ dst, float* __restrict__ cnt, int nE) {
    int e = blockIdx.x * blockDim.x + threadIdx.x;
    if (e < nE) atomicAdd(&cnt[dst[e]], 1.0f);
}

// wave-per-edge scatter-add: lane f covers features f, f+32, ...
__global__ void scatter_add(const float* __restrict__ x, const int* __restrict__ src,
                            const int* __restrict__ dst, float* __restrict__ msum,
                            int nE, int F) {
    int tid   = blockIdx.x * blockDim.x + threadIdx.x;
    int wave  = tid >> 5;
    int lane  = tid & 31;
    int waves = (gridDim.x * blockDim.x) >> 5;
    for (int e = wave; e < nE; e += waves) {
        int s = src[e], d = dst[e];
        const float* xr = x + (size_t)s * F;
        float* mr = msum + (size_t)d * F;
        for (int f = lane; f < F; f += 32) atomicAdd(&mr[f], xr[f]);
    }
}

// A1 = bf16[N,64]: [ msum/max(cnt,1) (27) | x (27) | zeros (10) ]
__global__ void build_A1(const float* __restrict__ msum, const float* __restrict__ cnt,
                         const float* __restrict__ x, unsigned short* __restrict__ A, int N) {
    long long total = (long long)N * 64;
    long long stride = (long long)gridDim.x * blockDim.x;
    for (long long idx = (long long)blockIdx.x * blockDim.x + threadIdx.x; idx < total; idx += stride) {
        int i = (int)(idx >> 6), c = (int)(idx & 63);
        float v;
        if (c < 27) {
            float rc = 1.0f / fmaxf(cnt[i], 1.0f);
            v = msum[(size_t)i * 27 + c] * rc;
        } else if (c < 54) {
            v = x[(size_t)i * 27 + (c - 27)];
        } else v = 0.0f;
        A[idx] = f2bf(v);
    }
}

// A2 = bf16[N,256]: [ msum/max(cnt,1) (128) | h (128) ]
__global__ void build_A2(const float* __restrict__ msum, const float* __restrict__ cnt,
                         const float* __restrict__ h, unsigned short* __restrict__ A, int N) {
    long long total = (long long)N * 256;
    long long stride = (long long)gridDim.x * blockDim.x;
    for (long long idx = (long long)blockIdx.x * blockDim.x + threadIdx.x; idx < total; idx += stride) {
        int i = (int)(idx >> 8), c = (int)(idx & 255);
        float v;
        if (c < 128) {
            float rc = 1.0f / fmaxf(cnt[i], 1.0f);
            v = msum[(size_t)i * 128 + c] * rc;
        } else {
            v = h[(size_t)i * 128 + (c - 128)];
        }
        A[idx] = f2bf(v);
    }
}

// Wc1 bf16[64,128] = [Wl(27) ; Wr(27) ; zeros(10)]
__global__ void build_W1(const float* __restrict__ Wl, const float* __restrict__ Wr,
                         unsigned short* __restrict__ W) {
    int idx = blockIdx.x * blockDim.x + threadIdx.x;
    if (idx >= 64 * 128) return;
    int r = idx / 128, c = idx % 128;
    float v = 0.0f;
    if (r < 27)      v = Wl[r * 128 + c];
    else if (r < 54) v = Wr[(r - 27) * 128 + c];
    W[idx] = f2bf(v);
}

// Wc2 bf16[256,64] = [Wl(128) ; Wr(128)]
__global__ void build_W2(const float* __restrict__ Wl, const float* __restrict__ Wr,
                         unsigned short* __restrict__ W) {
    int idx = blockIdx.x * blockDim.x + threadIdx.x;
    if (idx >= 256 * 64) return;
    int r = idx / 64, c = idx % 64;
    float v = (r < 128) ? Wl[r * 64 + c] : Wr[(r - 128) * 64 + c];
    W[idx] = f2bf(v);
}

// ---------------------------------------------------------------------------
// Pre-pack W (row-major KPAD x NC bf16) into per-lane WMMA B-fragments:
//   Bpack[((ct*KSTEPS + ks)*32 + lane)*16 + j] = W[(ks*32 + g*16 + j)*NC + ct*16 + n]
// so each wave's B fragment is one contiguous 32-byte load.
// ---------------------------------------------------------------------------
template <int KPAD, int NC>
__global__ void pack_Bfrag(const unsigned short* __restrict__ W,
                           unsigned short* __restrict__ Bp) {
    constexpr int KSTEPS = KPAD / 32;
    constexpr int CT     = NC / 16;
    int idx = blockIdx.x * blockDim.x + threadIdx.x;
    if (idx >= CT * KSTEPS * 32 * 16) return;
    int j    = idx & 15;
    int lane = (idx >> 4) & 31;
    int frag = idx >> 9;
    int ks   = frag % KSTEPS;
    int ct   = frag / KSTEPS;
    int g = lane >> 4, n = lane & 15;
    int k = ks * 32 + g * 16 + j;
    Bp[idx] = W[k * NC + ct * 16 + n];
}

// ---------------------------------------------------------------------------
// WMMA bf16 GEMM: out[M,NC] = act(A[M,KPAD] @ W[KPAD,NC] + bias)
//  block = 256 threads = 8 waves; wave -> 16x16 tiles.
//  CT col-tiles, RT row-sub-tiles per block pass (CT*RT == 8),
//  RPT row-tile iterations per wave with B fragments held in registers.
// ---------------------------------------------------------------------------
template <int KPAD, int NC, int CT, int RT, int RPT, bool RELU>
__global__ __launch_bounds__(256) void gemm_bf16(const unsigned short* __restrict__ A,
                                                 const unsigned short* __restrict__ Bp,
                                                 const float* __restrict__ bias,
                                                 float* __restrict__ out, int M) {
    constexpr int KSTEPS = KPAD / 32;
    int tid = threadIdx.x;
    int w = tid >> 5, lane = tid & 31;
    int rsub = w / CT, ct = w % CT;
    int g = lane >> 4;      // K-half selector
    int n = lane & 15;      // M row for A loads, N col for B/C

    // load all B fragments for this wave's column tile into registers
    v16bf bfr[KSTEPS];
#pragma unroll
    for (int ks = 0; ks < KSTEPS; ++ks) {
        const unsigned short* bp = Bp + (size_t)((ct * KSTEPS + ks) * 32 + lane) * 16;
        U32x8 bu;
        bu.lo = *reinterpret_cast<const uint4*>(bp);
        bu.hi = *reinterpret_cast<const uint4*>(bp + 8);
        bfr[ks] = __builtin_bit_cast(v16bf, bu);
    }

    int col = ct * 16 + n;
    float b = bias[col];
    int blockBase = blockIdx.x * (16 * RT * RPT);

#pragma unroll
    for (int t = 0; t < RPT; ++t) {
        int m0 = blockBase + (t * RT + rsub) * 16;
        if (m0 + 16 > M) break;

        v8f c = {0.f, 0.f, 0.f, 0.f, 0.f, 0.f, 0.f, 0.f};
        const unsigned short* arow = A + (size_t)(m0 + n) * KPAD;

#pragma unroll
        for (int ks = 0; ks < KSTEPS; ++ks) {
            int k0 = ks * 32;
            // A fragment: elems 0..15 <- K=k0+g*16+j
            U32x8 au;
            au.lo = *reinterpret_cast<const uint4*>(arow + k0 + g * 16);
            au.hi = *reinterpret_cast<const uint4*>(arow + k0 + g * 16 + 8);
            v16bf av = __builtin_bit_cast(v16bf, au);
            c = __builtin_amdgcn_wmma_f32_16x16x32_bf16(false, av, false, bfr[ks],
                                                        (short)0, c, false, false);
        }

#pragma unroll
        for (int r = 0; r < 8; ++r) {
            float v = c[r] + b;
            if (RELU) v = fmaxf(v, 0.0f);
            out[(size_t)(m0 + r + 8 * g) * NC + col] = v;
        }
    }
}

// ---------------------------------------------------------------------------
// d = sqrt(sum((z1-z2+1e-6)^2)); s = tanh(d * sign(pool_w))
// ---------------------------------------------------------------------------
__global__ void distance_score(const float* __restrict__ z1, const float* __restrict__ z2,
                               const float* __restrict__ poolw,
                               float* __restrict__ dval, float* __restrict__ sval, int N) {
    int i = blockIdx.x * blockDim.x + threadIdx.x;
    if (i >= N) return;
    const float* a = z1 + (size_t)i * 64;
    const float* b = z2 + (size_t)i * 64;
    float acc = 0.0f;
#pragma unroll 8
    for (int k = 0; k < 64; ++k) {
        float df = a[k] - b[k] + 1e-6f;
        acc += df * df;
    }
    float d = sqrtf(acc);
    float pw = poolw[0];
    float sgn = pw / sqrtf(pw * pw);
    dval[i] = d;
    sval[i] = tanhf(d * sgn);
}

// ---------------------------------------------------------------------------
// per-graph top-64 of 1000 scores (descending, ties -> lower index),
// pooled[b][j] = d[perm_j] * s[perm_j]
// ---------------------------------------------------------------------------
__global__ __launch_bounds__(256) void topk_pool(const float* __restrict__ sval,
                                                 const float* __restrict__ dval,
                                                 float* __restrict__ pooled) {
    const int NPG = 1000, K = 64;
    __shared__ float sv[NPG];
    __shared__ float rmax[256];
    __shared__ int   rarg[256];
    int gph = blockIdx.x, tid = threadIdx.x;
    for (int j = tid; j < NPG; j += 256) sv[j] = sval[gph * NPG + j];
    __syncthreads();
    for (int it = 0; it < K; ++it) {
        float best = -3.4e38f;
        int   barg = 0x7fffffff;
        for (int j = tid; j < NPG; j += 256) {
            float v = sv[j];
            if (v > best) { best = v; barg = j; }
        }
        rmax[tid] = best; rarg[tid] = barg;
        __syncthreads();
        for (int off = 128; off > 0; off >>= 1) {
            if (tid < off) {
                float vo = rmax[tid + off]; int ao = rarg[tid + off];
                if (vo > rmax[tid] || (vo == rmax[tid] && ao < rarg[tid])) {
                    rmax[tid] = vo; rarg[tid] = ao;
                }
            }
            __syncthreads();
        }
        if (tid == 0) {
            int a = rarg[0];
            pooled[gph * K + it] = dval[gph * NPG + a] * rmax[0];
            sv[a] = -3.4e38f;
        }
        __syncthreads();
    }
}

// ---------------------------------------------------------------------------
// MLP head on [B=100, 64]: fc+BN+relu, fc+BN+relu, fc+sigmoid. Single block.
// ---------------------------------------------------------------------------
__global__ __launch_bounds__(128) void mlp_head(const float* __restrict__ pooled,
                                                const float* __restrict__ l1W, const float* __restrict__ l1b,
                                                const float* __restrict__ g1, const float* __restrict__ b1,
                                                const float* __restrict__ l2W, const float* __restrict__ l2b,
                                                const float* __restrict__ g2, const float* __restrict__ b2,
                                                const float* __restrict__ l3W, const float* __restrict__ l3b,
                                                float* __restrict__ out, int B) {
    const float EPS = 1e-5f;
    __shared__ float xp[100 * 64];
    __shared__ float y1[100 * 32];
    __shared__ float y2[100 * 16];
    __shared__ float mu1[32], rs1[32];
    __shared__ float mu2[16], rs2[16];
    int tid = threadIdx.x;

    for (int i = tid; i < B * 64; i += 128) xp[i] = pooled[i];
    __syncthreads();

    for (int idx = tid; idx < B * 32; idx += 128) {
        int r = idx >> 5, c = idx & 31;
        float acc = l1b[c];
        for (int k = 0; k < 64; ++k) acc += xp[r * 64 + k] * l1W[k * 32 + c];
        y1[idx] = acc;
    }
    __syncthreads();
    if (tid < 32) {
        float m = 0.f;
        for (int r = 0; r < 100; ++r) m += y1[r * 32 + tid];
        m *= 0.01f;
        float v = 0.f;
        for (int r = 0; r < 100; ++r) { float d = y1[r * 32 + tid] - m; v += d * d; }
        v *= 0.01f;
        mu1[tid] = m; rs1[tid] = rsqrtf(v + EPS);
    }
    __syncthreads();
    for (int idx = tid; idx < B * 32; idx += 128) {
        int c = idx & 31;
        float v = (y1[idx] - mu1[c]) * rs1[c] * g1[c] + b1[c];
        y1[idx] = fmaxf(v, 0.0f);
    }
    __syncthreads();

    for (int idx = tid; idx < B * 16; idx += 128) {
        int r = idx >> 4, c = idx & 15;
        float acc = l2b[c];
        for (int k = 0; k < 32; ++k) acc += y1[r * 32 + k] * l2W[k * 16 + c];
        y2[idx] = acc;
    }
    __syncthreads();
    if (tid < 16) {
        float m = 0.f;
        for (int r = 0; r < 100; ++r) m += y2[r * 16 + tid];
        m *= 0.01f;
        float v = 0.f;
        for (int r = 0; r < 100; ++r) { float d = y2[r * 16 + tid] - m; v += d * d; }
        v *= 0.01f;
        mu2[tid] = m; rs2[tid] = rsqrtf(v + EPS);
    }
    __syncthreads();
    for (int idx = tid; idx < B * 16; idx += 128) {
        int c = idx & 15;
        float v = (y2[idx] - mu2[c]) * rs2[c] * g2[c] + b2[c];
        y2[idx] = fmaxf(v, 0.0f);
    }
    __syncthreads();

    if (tid < B) {
        float acc = l3b[0];
        for (int k = 0; k < 16; ++k) acc += y2[tid * 16 + k] * l3W[k];
        out[tid] = 1.0f / (1.0f + expf(-acc));
    }
}

// ---------------------------------------------------------------------------
// Host-side orchestration
// ---------------------------------------------------------------------------
extern "C" void kernel_launch(void* const* d_in, const int* in_sizes, int n_in,
                              void* d_out, int out_size, void* d_ws, size_t ws_size,
                              hipStream_t stream) {
    const float* x1   = (const float*)d_in[0];
    const float* x2   = (const float*)d_in[1];
    const int*   es1  = (const int*)d_in[2];
    const int*   ed1  = (const int*)d_in[3];
    const int*   es2  = (const int*)d_in[4];
    const int*   ed2  = (const int*)d_in[5];
    const float* c1Wl = (const float*)d_in[6];
    const float* c1bl = (const float*)d_in[7];
    const float* c1Wr = (const float*)d_in[8];
    const float* c2Wl = (const float*)d_in[9];
    const float* c2bl = (const float*)d_in[10];
    const float* c2Wr = (const float*)d_in[11];
    const float* poolw= (const float*)d_in[12];
    const float* l1W  = (const float*)d_in[13];
    const float* l1b  = (const float*)d_in[14];
    const float* bn1g = (const float*)d_in[15];
    const float* bn1b = (const float*)d_in[16];
    const float* l2W  = (const float*)d_in[17];
    const float* l2b  = (const float*)d_in[18];
    const float* bn2g = (const float*)d_in[19];
    const float* bn2b = (const float*)d_in[20];
    const float* l3W  = (const float*)d_in[21];
    const float* l3b  = (const float*)d_in[22];

    const int N = in_sizes[0] / 27;   // 100000
    const int E = in_sizes[2];        // 1600000
    const int B = N / 1000;           // 100

    // bump allocator over d_ws (256B aligned)
    char* wp = (char*)d_ws;
    auto carve = [&](size_t bytes) -> void* {
        void* r = (void*)wp;
        wp += (bytes + 255) & ~(size_t)255;
        return r;
    };
    float*          z1     = (float*)carve((size_t)N * 64 * 4);
    float*          z2     = (float*)carve((size_t)N * 64 * 4);
    float*          dval   = (float*)carve((size_t)N * 4);
    float*          sval   = (float*)carve((size_t)N * 4);
    float*          pooled = (float*)carve((size_t)B * 64 * 4);
    unsigned short* Wc1    = (unsigned short*)carve(64 * 128 * 2);
    unsigned short* Wc2    = (unsigned short*)carve(256 * 64 * 2);
    unsigned short* Bp1    = (unsigned short*)carve(64 * 128 * 2);   // packed fragments
    unsigned short* Bp2    = (unsigned short*)carve(256 * 64 * 2);
    float*          msum   = (float*)carve((size_t)N * 128 * 4);
    float*          cnt    = (float*)carve((size_t)N * 4);
    unsigned short* Abuf   = (unsigned short*)carve((size_t)N * 256 * 2);
    float*          hbuf   = (float*)carve((size_t)N * 128 * 4);

    build_W1<<<(64 * 128 + 255) / 256, 256, 0, stream>>>(c1Wl, c1Wr, Wc1);
    build_W2<<<(256 * 64 + 255) / 256, 256, 0, stream>>>(c2Wl, c2Wr, Wc2);
    pack_Bfrag<64, 128><<<(64 * 128 + 255) / 256, 256, 0, stream>>>(Wc1, Bp1);
    pack_Bfrag<256, 64><<<(256 * 64 + 255) / 256, 256, 0, stream>>>(Wc2, Bp2);

    for (int pass = 0; pass < 2; ++pass) {
        const float* x   = pass ? x2  : x1;
        const int*   src = pass ? es2 : es1;
        const int*   dst = pass ? ed2 : ed1;
        float*       z   = pass ? z2  : z1;

        // ---- layer 1: mean-aggregate (F=27) + fused [mean|x] GEMM, relu ----
        zero_f32<<<4096, 256, 0, stream>>>(msum, (long long)N * 27);
        zero_f32<<<256, 256, 0, stream>>>(cnt, N);
        count_edges<<<(E + 255) / 256, 256, 0, stream>>>(dst, cnt, E);
        scatter_add<<<8192, 256, 0, stream>>>(x, src, dst, msum, E, 27);
        build_A1<<<8192, 256, 0, stream>>>(msum, cnt, x, Abuf, N);
        // 16 rows/tile * RT=1 * RPT=4 = 64 rows per block
        gemm_bf16<64, 128, 8, 1, 4, true>
            <<<(N + 63) / 64, 256, 0, stream>>>(Abuf, Bp1, c1bl, hbuf, N);

        // ---- layer 2: mean-aggregate (F=128) + fused [mean|h] GEMM ----
        zero_f32<<<8192, 256, 0, stream>>>(msum, (long long)N * 128);
        scatter_add<<<8192, 256, 0, stream>>>(hbuf, src, dst, msum, E, 128);
        build_A2<<<8192, 256, 0, stream>>>(msum, cnt, hbuf, Abuf, N);
        // 16 rows/tile * RT=2 * RPT=4 = 128 rows per block
        gemm_bf16<256, 64, 4, 2, 4, false>
            <<<(N + 127) / 128, 256, 0, stream>>>(Abuf, Bp2, c2bl, z, N);
    }

    distance_score<<<(N + 255) / 256, 256, 0, stream>>>(z1, z2, poolw, dval, sval, N);
    topk_pool<<<B, 256, 0, stream>>>(sval, dval, pooled);
    mlp_head<<<1, 128, 0, stream>>>(pooled, l1W, l1b, bn1g, bn1b,
                                    l2W, l2b, bn2g, bn2b, l3W, l3b,
                                    (float*)d_out, B);
}
